// GAT_40140764348987
// MI455X (gfx1250) — matrix operand order repaired and checked by reading, not compiled, not run on hardware
//
#include <hip/hip_runtime.h>
#include <hip/hip_bf16.h>

// GATv2 2-layer + mean-pool + linear head for MI455X (gfx1250, wave32).
// Bandwidth-bound (~7 GB edge traffic -> ~0.3 ms @ 23.3 TB/s); GEMMs (~16 GFLOP)
// run on v_wmma_f32_16x16x32_f16 with TDM-staged activation tiles and
// fragment-order pre-packed f16 weights (2x global_load_b128 per B frag).

#define N_NODES 50000
#define N_EDGES 800000
#define N_GRAPHS 1000
#define HID 256
#define HEADS 4
#define CPH 64

typedef __attribute__((ext_vector_type(16))) _Float16 v16h;
typedef __attribute__((ext_vector_type(8)))  float    v8f;
typedef __attribute__((ext_vector_type(4)))  unsigned v4u;
typedef __attribute__((ext_vector_type(8)))  int      v8i;
typedef __attribute__((ext_vector_type(4)))  int      v4i;

#if defined(__has_builtin)
#if __has_builtin(__builtin_amdgcn_tensor_load_to_lds) && \
    __has_builtin(__builtin_amdgcn_s_wait_tensorcnt)
#define USE_TDM 1
#endif
#endif
#ifndef USE_TDM
#define USE_TDM 0
#endif

// ---------------- order-preserving float<->uint keys for atomicMax ----------
__device__ __forceinline__ unsigned enc_f32(float f) {
    unsigned u = __float_as_uint(f);
    return (u & 0x80000000u) ? ~u : (u | 0x80000000u);
}
__device__ __forceinline__ float dec_f32(unsigned k) {
    unsigned u = (k & 0x80000000u) ? (k & 0x7FFFFFFFu) : ~k;
    return __uint_as_float(u);
}
// enc(-inf) == ~0xFF800000 == 0x007FFFFF
#define ENC_NEG_INF 0x007FFFFFu

// ---------------- fill kernels ---------------------------------------------
__global__ void gat_fill_f32(float* __restrict__ p, float v, long long n) {
    long long i = (long long)blockIdx.x * blockDim.x + threadIdx.x;
    long long stride = (long long)gridDim.x * blockDim.x;
    for (; i < n; i += stride) p[i] = v;
}
__global__ void gat_fill_u32(unsigned* __restrict__ p, unsigned v, long long n) {
    long long i = (long long)blockIdx.x * blockDim.x + threadIdx.x;
    long long stride = (long long)gridDim.x * blockDim.x;
    for (; i < n; i += stride) p[i] = v;
}

// ---------------- weight pre-pack into WMMA B-fragment order ----------------
// Packed linear index: (((kc*16 + ntile)*32 + lane)*16 + j)
//   k   = kc*32 + (lane>>4)*16 + j   (B: lanes 0-15 hold K k0..k0+15, 16-31 next 16)
//   col = ntile*16 + (lane&15)
__global__ __launch_bounds__(256) void gat_pack_w(
    const float* __restrict__ W, _Float16* __restrict__ WP, int K)
{
    long long i = (long long)blockIdx.x * blockDim.x + threadIdx.x;
    if (i >= (long long)K * HID) return;
    const int j  = (int)(i & 15);
    const int l  = (int)((i >> 4) & 31);
    const int nt = (int)((i >> 9) & 15);
    const int kc = (int)(i >> 13);
    const int k   = kc * 32 + ((l >> 4) << 4) + j;
    const int col = (nt << 4) + (l & 15);
    WP[i] = (_Float16)W[(long long)k * HID + col];
}

// ---------------- dual GEMM: XL = X*Wl + bl ; XR = X*Wr + br ----------------
// One block = 16 output rows. 8 waves; each wave owns 4 of the 32 (ntile,matrix)
// 16x16 tiles. X tile DMA'd to LDS by the TDM (f32), converted to f16 once.
#define KPAD 264   // 256 max K + 8 halves pad

__global__ __launch_bounds__(256) void gat_gemm_dual(
    const float* __restrict__ X,
    const v16h* __restrict__ WPl, const float* __restrict__ bl,
    const v16h* __restrict__ WPr, const float* __restrict__ br,
    float* __restrict__ XL, float* __restrict__ XR,
    int K, int Nrows)
{
    __shared__ _Float16 Xs[16 * KPAD];
    const int tid  = threadIdx.x;
    const int row0 = blockIdx.x * 16;

#if USE_TDM
    __shared__ float Xf[16 * 256];
    if (tid < 32) {
        // Tensor DMA descriptor: 2D tile, 16 rows x K cols, 4-byte elements.
        const unsigned lds = (unsigned)(unsigned long long)(const void*)&Xf[0];
        const unsigned long long ga =
            (unsigned long long)(const void*)(X + (long long)row0 * K);
        const unsigned td0 = (unsigned)K;                 // tensor_dim0
        const unsigned td1 = (unsigned)(Nrows - row0);    // tensor_dim1 (OOB rows -> 0)
        v4u g0; v8i g1; v4i g2 = {}; v4i g3 = {}; v8i g4 = {};
        g0[0] = 1u;                                        // count=1, user mode
        g0[1] = lds;                                       // lds_addr
        g0[2] = (unsigned)(ga & 0xFFFFFFFFull);            // global_addr[31:0]
        g0[3] = (unsigned)((ga >> 32) & 0x01FFFFFFull)     // global_addr[56:32]
              | (2u << 30);                                // type=2 ("image")
        g1[0] = (int)(2u << 16);                           // data_size=2 (4 bytes)
        g1[1] = (int)((td0 & 0xFFFFu) << 16);              // tensor_dim0[15:0]
        g1[2] = (int)((td0 >> 16) | ((td1 & 0xFFFFu) << 16)); // td0[31:16] | td1[15:0]
        g1[3] = (int)((td1 >> 16) | ((unsigned)K << 16));  // td1[31:16] | tile_dim0=K
        g1[4] = 16;                                        // tile_dim1=16, tile_dim2=0
        g1[5] = (int)(unsigned)K;                          // tensor_dim0_stride[31:0]
        g1[6] = 0;                                         // stride[47:32], dim1_stride lo
        g1[7] = 0;
        __builtin_amdgcn_tensor_load_to_lds(g0, g1, g2, g3, g4, 0);
        __builtin_amdgcn_s_wait_tensorcnt(0);
    }
    __syncthreads();
    for (int idx = tid; idx < 16 * K; idx += 256) {
        int r = idx / K, c = idx - r * K;
        Xs[r * KPAD + c] = (_Float16)Xf[idx];
    }
#else
    for (int idx = tid; idx < 16 * K; idx += 256) {
        int r = idx / K, c = idx - r * K;
        int gr = row0 + r;
        float v = (gr < Nrows) ? X[(long long)gr * K + c] : 0.0f;
        Xs[r * KPAD + c] = (_Float16)v;
    }
#endif
    __syncthreads();

    const int wave = tid >> 5;
    const int lane = tid & 31;
    const int m    = lane & 15;   // row (A) / col (B,C)
    const int grp  = lane >> 4;

    v8f acc[4] = {};

    for (int k0 = 0; k0 < K; k0 += 32) {
        // A fragment: 16x32 f16, contiguous K (lanes 0-15: K k0+0..15;
        // lanes 16-31: K k0+16..31), row M = lane&15.
        v16h a;
        const int abase = k0 + grp * 16;
        #pragma unroll
        for (int j = 0; j < 8; ++j) {
            a[j]     = Xs[m * KPAD + abase + j];
            a[j + 8] = Xs[m * KPAD + abase + 8 + j];
        }
        const long long kc = k0 >> 5;
        #pragma unroll
        for (int t = 0; t < 4; ++t) {
            const int tile = wave * 4 + t;
            const v16h* __restrict__ WP = (tile < 16) ? WPl : WPr;
            const int nt = tile & 15;
            v16h b = WP[(kc * 16 + nt) * 32 + lane];
            acc[t] = __builtin_amdgcn_wmma_f32_16x16x32_f16(
                false, a, false, b, (short)0, acc[t], false, false);
        }
    }

    #pragma unroll
    for (int t = 0; t < 4; ++t) {
        const int tile = wave * 4 + t;
        float* __restrict__ OUT       = (tile < 16) ? XL : XR;
        const float* __restrict__ bia = (tile < 16) ? bl : br;
        const int col = ((tile & 15) << 4) + m;
        const float bv = bia[col];
        #pragma unroll
        for (int r = 0; r < 8; ++r) {
            int row = row0 + r + grp * 8;   // C/D: lanes 16-31 hold M = r+8
            if (row < Nrows)
                OUT[(long long)row * HID + col] = acc[t][r] + bv;
        }
    }
}

// ---------------- edge pass 1: scores + segment max ------------------------
// One wave per edge; lane handles 8 channels (2x float4); head = lane>>3.
__global__ __launch_bounds__(256) void gat_edge_score(
    const float* __restrict__ XL, const float* __restrict__ XR,
    const int* __restrict__ src, const int* __restrict__ dst,
    const float* __restrict__ att,
    float* __restrict__ score, unsigned* __restrict__ mkey, int E)
{
    const int e = blockIdx.x * 8 + (threadIdx.x >> 5);
    const int lane = threadIdx.x & 31;
    if (e >= E) return;
    const int s = src[e], d = dst[e];
    const int ch = lane * 8;
    const int h  = lane >> 3;
    const float4* __restrict__ pl = (const float4*)(XL + (long long)s * HID + ch);
    const float4* __restrict__ pr = (const float4*)(XR + (long long)d * HID + ch);
    float acc = 0.0f;
    #pragma unroll
    for (int q = 0; q < 2; ++q) {
        float4 a = pl[q], b = pr[q];
        float v[4] = {a.x + b.x, a.y + b.y, a.z + b.z, a.w + b.w};
        #pragma unroll
        for (int j = 0; j < 4; ++j) {
            float t = v[j];
            t = (t > 0.0f) ? t : 0.2f * t;          // leaky_relu(0.2)
            acc += t * att[h * CPH + ((ch + q * 4 + j) & (CPH - 1))];
        }
    }
    acc += __shfl_xor(acc, 1);
    acc += __shfl_xor(acc, 2);
    acc += __shfl_xor(acc, 4);
    if ((lane & 7) == 0) {
        score[(long long)e * HEADS + h] = acc;
        atomicMax(&mkey[(long long)d * HEADS + h], enc_f32(acc));
    }
}

// ---------------- edge pass 2: exp + segment sum ---------------------------
__global__ __launch_bounds__(256) void gat_edge_exp(
    const int* __restrict__ dst, float* __restrict__ score,
    const unsigned* __restrict__ mkey, float* __restrict__ denom, long long EH)
{
    long long i = (long long)blockIdx.x * blockDim.x + threadIdx.x;
    if (i >= EH) return;
    const int e = (int)(i >> 2);
    const int h = (int)(i & 3);
    const int d = dst[e];
    const float m  = dec_f32(mkey[(long long)d * HEADS + h]);
    const float ex = __expf(score[i] - m);
    score[i] = ex;
    atomicAdd(&denom[(long long)d * HEADS + h], ex);
}

// ---------------- edge pass 3: alpha * xl[src] -> H[dst] -------------------
__global__ __launch_bounds__(256) void gat_edge_aggr(
    const float* __restrict__ XL,
    const int* __restrict__ src, const int* __restrict__ dst,
    const float* __restrict__ ex, const float* __restrict__ denom,
    float* __restrict__ Hout, int E)
{
    const int e = blockIdx.x * 8 + (threadIdx.x >> 5);
    const int lane = threadIdx.x & 31;
    if (e >= E) return;
    const int s = src[e], d = dst[e];
    const int h = lane >> 3;
    const float alpha = ex[(long long)e * HEADS + h] /
                        (denom[(long long)d * HEADS + h] + 1e-16f);
    const int ch = lane * 8;
    const float4* __restrict__ pl = (const float4*)(XL + (long long)s * HID + ch);
    float* __restrict__ out = Hout + (long long)d * HID + ch;
    #pragma unroll
    for (int q = 0; q < 2; ++q) {
        float4 a = pl[q];
        atomicAdd(out + q * 4 + 0, alpha * a.x);
        atomicAdd(out + q * 4 + 1, alpha * a.y);
        atomicAdd(out + q * 4 + 2, alpha * a.z);
        atomicAdd(out + q * 4 + 3, alpha * a.w);
    }
}

// ---------------- bias + relu ----------------------------------------------
__global__ __launch_bounds__(256) void gat_bias_relu(
    float* __restrict__ Hp, const float* __restrict__ bias, long long n)
{
    long long i = (long long)blockIdx.x * blockDim.x + threadIdx.x;
    if (i >= n) return;
    float v = Hp[i] + bias[(int)(i & (HID - 1))];
    Hp[i] = (v > 0.0f) ? v : 0.0f;
}

// ---------------- mean pool (atomic) ---------------------------------------
__global__ __launch_bounds__(256) void gat_pool(
    const float* __restrict__ Hp, const int* __restrict__ batch,
    float* __restrict__ pool, float* __restrict__ cnt, int N)
{
    long long i = (long long)blockIdx.x * blockDim.x + threadIdx.x;
    if (i >= (long long)N * HID) return;
    const int n = (int)(i >> 8);
    const int c = (int)(i & (HID - 1));
    const int g = batch[n];
    atomicAdd(&pool[(long long)g * HID + c], Hp[i]);
    if (c == 0) atomicAdd(&cnt[g], 1.0f);
}

// ---------------- head: out[g] = dot(pool[g]/cnt, Wffn) + bffn -------------
__global__ __launch_bounds__(256) void gat_head(
    const float* __restrict__ pool, const float* __restrict__ cnt,
    const float* __restrict__ Wffn, const float* __restrict__ bffn,
    float* __restrict__ out)
{
    __shared__ float red[8];
    const int g = blockIdx.x;
    const int t = threadIdx.x;
    float c = cnt[g];
    c = (c > 1.0f) ? c : 1.0f;
    float v = pool[(long long)g * HID + t] / c * Wffn[t];
    v += __shfl_xor(v, 1);
    v += __shfl_xor(v, 2);
    v += __shfl_xor(v, 4);
    v += __shfl_xor(v, 8);
    v += __shfl_xor(v, 16);
    if ((t & 31) == 0) red[t >> 5] = v;
    __syncthreads();
    if (t < 8) {
        float r = red[t];
        r += __shfl_xor(r, 1);
        r += __shfl_xor(r, 2);
        r += __shfl_xor(r, 4);
        if (t == 0) out[g] = r + bffn[0];
    }
}

// ---------------------------------------------------------------------------
extern "C" void kernel_launch(void* const* d_in, const int* in_sizes, int n_in,
                              void* d_out, int out_size, void* d_ws, size_t ws_size,
                              hipStream_t stream) {
    (void)in_sizes; (void)n_in; (void)out_size; (void)ws_size;
    const int N = N_NODES, E = N_EDGES, G = N_GRAPHS;

    const float* x     = (const float*)d_in[0];
    const int*   ei    = (const int*)d_in[1];   // [2,E] (int32 per JAX default)
    const int*   batch = (const int*)d_in[2];
    const float* W1l   = (const float*)d_in[3];
    const float* b1l   = (const float*)d_in[4];
    const float* W1r   = (const float*)d_in[5];
    const float* b1r   = (const float*)d_in[6];
    const float* att1  = (const float*)d_in[7];
    const float* bias1 = (const float*)d_in[8];
    const float* W2l   = (const float*)d_in[9];
    const float* b2l   = (const float*)d_in[10];
    const float* W2r   = (const float*)d_in[11];
    const float* b2r   = (const float*)d_in[12];
    const float* att2  = (const float*)d_in[13];
    const float* bias2 = (const float*)d_in[14];
    const float* Wffn  = (const float*)d_in[15];
    const float* bffn  = (const float*)d_in[16];

    const int* src = ei;
    const int* dst = ei + E;

    // workspace layout (256B aligned)
    char* base = (char*)d_ws;
    size_t off = 0;
    auto take = [&](size_t bytes) -> void* {
        void* p = base + off;
        off = (off + bytes + 255) & ~(size_t)255;
        return p;
    };
    float*    XL   = (float*)take((size_t)N * HID * 4);
    float*    XR   = (float*)take((size_t)N * HID * 4);
    float*    Hb   = (float*)take((size_t)N * HID * 4);
    float*    SC   = (float*)take((size_t)E * HEADS * 4);
    unsigned* MK   = (unsigned*)take((size_t)N * HEADS * 4);
    float*    DEN  = (float*)take((size_t)N * HEADS * 4);
    float*    POOL = (float*)take((size_t)G * HID * 4);
    float*    CNT  = (float*)take((size_t)G * 4);
    _Float16* WP1l = (_Float16*)take((size_t)64  * HID * 2);
    _Float16* WP1r = (_Float16*)take((size_t)64  * HID * 2);
    _Float16* WP2l = (_Float16*)take((size_t)HID * HID * 2);
    _Float16* WP2r = (_Float16*)take((size_t)HID * HID * 2);

    // pack all weights into B-fragment order (f16), once per launch
    gat_pack_w<<<64,  256, 0, stream>>>(W1l, WP1l, 64);
    gat_pack_w<<<64,  256, 0, stream>>>(W1r, WP1r, 64);
    gat_pack_w<<<256, 256, 0, stream>>>(W2l, WP2l, HID);
    gat_pack_w<<<256, 256, 0, stream>>>(W2r, WP2r, HID);

    auto run_layer = [&](const float* in, int K,
                         const _Float16* WPl, const float* bl,
                         const _Float16* WPr, const float* br,
                         const float* att, const float* bias) {
        gat_gemm_dual<<<(N + 15) / 16, 256, 0, stream>>>(
            in, (const v16h*)WPl, bl, (const v16h*)WPr, br, XL, XR, K, N);
        gat_fill_f32<<<2048, 256, 0, stream>>>(Hb, 0.0f, (long long)N * HID);
        gat_fill_f32<<<256, 256, 0, stream>>>(DEN, 0.0f, (long long)N * HEADS);
        gat_fill_u32<<<256, 256, 0, stream>>>(MK, ENC_NEG_INF, (long long)N * HEADS);
        gat_edge_score<<<(E + 7) / 8, 256, 0, stream>>>(XL, XR, src, dst, att,
                                                        SC, MK, E);
        gat_edge_exp<<<(int)(((long long)E * HEADS + 255) / 256), 256, 0, stream>>>(
            dst, SC, MK, DEN, (long long)E * HEADS);
        gat_edge_aggr<<<(E + 7) / 8, 256, 0, stream>>>(XL, src, dst, SC, DEN, Hb, E);
        gat_bias_relu<<<(int)(((long long)N * HID + 255) / 256), 256, 0, stream>>>(
            Hb, bias, (long long)N * HID);
    };

    run_layer(x, 64, WP1l, b1l, WP1r, b1r, att1, bias1);
    run_layer(Hb, HID, WP2l, b2l, WP2r, b2r, att2, bias2);

    gat_fill_f32<<<64, 256, 0, stream>>>(POOL, 0.0f, (long long)G * HID);
    gat_fill_f32<<<4, 256, 0, stream>>>(CNT, 0.0f, (long long)G);
    gat_pool<<<(int)(((long long)N * HID + 255) / 256), 256, 0, stream>>>(
        Hb, batch, POOL, CNT, N);
    gat_head<<<G, 256, 0, stream>>>(POOL, CNT, Wffn, bffn, (float*)d_out);
}
